// Mamba_28552942584210
// MI455X (gfx1250) — compile-verified, matrix operand order checked by reading, not run on hardware
//
#include <hip/hip_runtime.h>
#include <hip/hip_bf16.h>

// ---- problem dims (fixed by reference) ----
#define D_MODEL 1024
#define D_STATE 16
#define D_CONV  4
#define DT_RANK 64
#define D_INNER 2048
#define BATCH   4
#define SEQ     4096
#define NTOK    (BATCH*SEQ)        // 16384
#define XPROJ_N 96                 // DT_RANK + 2*D_STATE
#define XPROJ_NP 128               // padded to a full N tile
#define NCH     16                 // scan chunks along L
#define CHL     (SEQ/NCH)          // 256 steps per chunk

typedef __attribute__((ext_vector_type(16))) __bf16 bf16x16;
typedef __attribute__((ext_vector_type(8)))  __bf16 bf16x8;
typedef __attribute__((ext_vector_type(8)))  float  f32x8;

union BFrag { bf16x16 v; struct { bf16x8 lo, hi; } h; };

// ================= WMMA GEMM: C[M,N] = A[M,K] * W[N,K]^T ==================
// A row-major (M,K) bf16; W row-major (N,K) bf16 (PyTorch-style weight);
// C row-major (M,N) f32.  M%128==0, N%128==0, K%64==0.
// Tiles are staged with GLOBAL_LOAD_ASYNC_TO_LDS_B128 (ASYNCcnt DMA) into a
// double-buffered LDS, overlapping the next K-tile's fill with WMMA compute.
#define TM 128
#define TN 128
#define KT 64
#define LROW 88   // padded LDS row: 176B = 16B-aligned rows, 16 rows hit
                  // distinct bank groups (r*44 mod 64 all distinct)

__global__ __launch_bounds__(256) void k_gemm_bf16_wmma(
    const __bf16* __restrict__ A, const __bf16* __restrict__ Bw,
    float* __restrict__ C, int M, int N, int K)
{
  __shared__ __bf16 As[2][TM * LROW];
  __shared__ __bf16 Bs[2][TN * LROW];
  const int tid  = threadIdx.x;
  const int wave = tid >> 5;            // 0..7  -> 2x4 wave grid
  const int lane = tid & 31;
  const int wM   = (wave >> 2) * 64;    // wave M offset in block tile
  const int wN   = (wave & 3)  * 32;    // wave N offset in block tile
  const int row  = lane & 15;           // fragment row (M for A, N for B)
  const int kh   = lane >> 4;           // K-half select per ISA 16-bit layout

  f32x8 acc[4][2] = {};                 // 64x32 output per wave

  // async copy mapping: each thread moves half a 64-col row (4x 16B) per matrix
  const int lr = tid >> 1;
  const int lc = (tid & 1) * 32;
  const __bf16* Ag = A  + (size_t)(blockIdx.y * TM + lr) * K + lc;
  const __bf16* Bg = Bw + (size_t)(blockIdx.x * TN + lr) * K + lc;
  // wave-relative LDS byte addresses of this thread's write slots
  const unsigned laBase[2] = {
      (unsigned)(size_t)&As[0][lr * LROW + lc],
      (unsigned)(size_t)&As[1][lr * LROW + lc] };
  const unsigned lbBase[2] = {
      (unsigned)(size_t)&Bs[0][lr * LROW + lc],
      (unsigned)(size_t)&Bs[1][lr * LROW + lc] };

  auto stage = [&](int buf, int kof) {
    const __bf16* ga = Ag + kof;
    const __bf16* gb = Bg + kof;
    #pragma unroll
    for (int u = 0; u < 4; ++u) {
      asm volatile("global_load_async_to_lds_b128 %0, %1, off"
                   :: "v"(laBase[buf] + u * 16), "v"(ga + u * 8) : "memory");
      asm volatile("global_load_async_to_lds_b128 %0, %1, off"
                   :: "v"(lbBase[buf] + u * 16), "v"(gb + u * 8) : "memory");
    }
  };

  const int nt = K / KT;
  stage(0, 0);
  for (int i = 0; i < nt; ++i) {
    if (i + 1 < nt) {
      stage((i + 1) & 1, (i + 1) * KT);          // prefetch next tile
      // async loads complete in order: <=8 outstanding => tile i landed
      asm volatile("s_wait_asynccnt 0x8" ::: "memory");
    } else {
      asm volatile("s_wait_asynccnt 0x0" ::: "memory");
    }
    __syncthreads();

    const __bf16* AsB = As[i & 1];
    const __bf16* BsB = Bs[i & 1];
    #pragma unroll
    for (int kk = 0; kk < KT; kk += 32) {
      // B fragments (reused across the 4 M tiles)
      BFrag bf[2];
      #pragma unroll
      for (int j = 0; j < 2; ++j) {
        const __bf16* p = &BsB[(wN + j * 16 + row) * LROW + kk + kh * 8];
        bf[j].h.lo = *(const bf16x8*)p;
        bf[j].h.hi = *(const bf16x8*)(p + 16);   // cols +16
      }
      #pragma unroll
      for (int i4 = 0; i4 < 4; ++i4) {
        BFrag af;
        const __bf16* p = &AsB[(wM + i4 * 16 + row) * LROW + kk + kh * 8];
        af.h.lo = *(const bf16x8*)p;
        af.h.hi = *(const bf16x8*)(p + 16);
        #pragma unroll
        for (int j = 0; j < 2; ++j)
          acc[i4][j] = __builtin_amdgcn_wmma_f32_16x16x32_bf16(
              false, af.v, false, bf[j].v, (short)0, acc[i4][j], false, false);
      }
    }
    __syncthreads();   // all waves done reading buf before it is refilled
  }

  // C/D layout: VGPR r -> M = r + 8*(lane>=16), N = lane&15
  #pragma unroll
  for (int i = 0; i < 4; ++i)
    #pragma unroll
    for (int j = 0; j < 2; ++j) {
      const int gm = blockIdx.y * TM + wM + i * 16 + kh * 8;
      const int gn = blockIdx.x * TN + wN + j * 16 + row;
      float* cp = C + (size_t)gm * N + gn;
      #pragma unroll
      for (int r = 0; r < 8; ++r) cp[(size_t)r * N] = acc[i][j][r];
    }
}

// ================= elementwise / conversion kernels ==================
__global__ __launch_bounds__(256) void k_f32_to_bf16(
    const float* __restrict__ in, __bf16* __restrict__ out, size_t n)
{
  size_t i = (size_t)blockIdx.x * 256 + threadIdx.x;
  if (i < n) out[i] = (__bf16)in[i];
}

// x_proj_w (96,2048) -> bf16 padded to (128,2048) with zero rows
__global__ __launch_bounds__(256) void k_cvt_pad_xproj(
    const float* __restrict__ w, __bf16* __restrict__ out)
{
  size_t gid = (size_t)blockIdx.x * 256 + threadIdx.x;   // 128*2048
  int r = (int)(gid / D_INNER);
  int c = (int)(gid % D_INNER);
  float v = (r < XPROJ_N) ? w[(size_t)r * D_INNER + c] : 0.f;
  out[gid] = (__bf16)v;
}

// causal depthwise conv1d + bias + SiLU over x-half of xz, emit bf16 x
__global__ __launch_bounds__(256) void k_conv_silu(
    const float* __restrict__ xz, const float* __restrict__ cw,
    const float* __restrict__ cb, __bf16* __restrict__ x_bf)
{
  size_t gid = (size_t)blockIdx.x * 256 + threadIdx.x;   // NTOK*D_INNER
  int d = (int)(gid % D_INNER);
  size_t t = gid / D_INNER;
  int l = (int)(t % SEQ);
  size_t brow = t - (size_t)l;                           // b*SEQ
  float acc = cb[d];
  #pragma unroll
  for (int j = 0; j < D_CONV; ++j) {
    int ll = l + j - (D_CONV - 1);
    if (ll >= 0)
      acc = fmaf(cw[d * D_CONV + j],
                 xz[(brow + (size_t)ll) * (size_t)(2 * D_INNER) + d], acc);
  }
  acc = acc / (1.f + __expf(-acc));                      // SiLU
  x_bf[gid] = (__bf16)acc;
}

// dt_in = x_dbl[:, 0:64] -> bf16
__global__ __launch_bounds__(256) void k_extract_dtin(
    const float* __restrict__ xdbl, __bf16* __restrict__ out)
{
  size_t gid = (size_t)blockIdx.x * 256 + threadIdx.x;   // NTOK*DT_RANK
  size_t t = gid >> 6;
  int c = (int)(gid & 63);
  out[gid] = (__bf16)xdbl[t * XPROJ_NP + c];
}

__device__ __forceinline__ float softplusf(float v) {
  return (v > 20.f) ? v : log1pf(__expf(v));
}

// ============== chunked selective scan: 3 passes over L-chunks ==============
// Recurrence h <- h*exp(dt*A) + dt*x*B is diagonal-linear; a chunk's total
// decay is exp(A * sum(dt)), so chunk summaries are (sum_dt, h_end).
//
// Pass 1: per (b,d,chunk) scan from h=0; emit sum_dt and h_end[16].
__global__ __launch_bounds__(256) void k_scan_pass1(
    const __bf16* __restrict__ x_bf,    // (NTOK, D_INNER)
    const float*  __restrict__ x_dbl,   // (NTOK, 128): B at col 64..79
    const float*  __restrict__ dt_raw,  // (NTOK, D_INNER)
    const float*  __restrict__ dt_bias, // (D_INNER)
    const float*  __restrict__ A_log,   // (D_INNER, D_STATE)
    float*        __restrict__ hEnd,    // (BATCH*NCH*D_INNER, 16)
    float*        __restrict__ dtSum)   // (BATCH*NCH*D_INNER)
{
  const int d = blockIdx.x * 256 + threadIdx.x;
  const int c = blockIdx.y;
  const int b = blockIdx.z;

  float Areg[D_STATE];
  #pragma unroll
  for (int n = 0; n < D_STATE; ++n) Areg[n] = -__expf(A_log[d * D_STATE + n]);
  const float bias = dt_bias[d];

  float h[D_STATE] = {};
  float ds = 0.f;
  for (int l = c * CHL; l < (c + 1) * CHL; ++l) {
    const size_t t = (size_t)b * SEQ + l;
    float dt = softplusf(dt_raw[t * D_INNER + d] + bias);
    ds += dt;
    const float dx = dt * (float)x_bf[t * D_INNER + d];
    const float* Bv = x_dbl + t * XPROJ_NP + 64;   // broadcast across lanes
    #pragma unroll
    for (int n = 0; n < D_STATE; ++n)
      h[n] = fmaf(h[n], __expf(dt * Areg[n]), dx * Bv[n]);
  }
  const size_t base = ((size_t)b * NCH + c) * D_INNER + d;
  dtSum[base] = ds;
  #pragma unroll
  for (int n = 0; n < D_STATE; ++n) hEnd[base * D_STATE + n] = h[n];
}

// Pass 2: per (b,d), sequential fix-up over NCH chunk boundaries.
__global__ __launch_bounds__(256) void k_scan_pass2(
    const float* __restrict__ A_log,
    const float* __restrict__ hEnd,
    const float* __restrict__ dtSum,
    float*       __restrict__ hIn)      // (BATCH*NCH*D_INNER, 16)
{
  const int gid = blockIdx.x * 256 + threadIdx.x;   // BATCH*D_INNER
  const int d = gid % D_INNER;
  const int b = gid / D_INNER;

  float Areg[D_STATE];
  #pragma unroll
  for (int n = 0; n < D_STATE; ++n) Areg[n] = -__expf(A_log[d * D_STATE + n]);

  float h[D_STATE] = {};
  for (int c = 0; c < NCH; ++c) {
    const size_t base = ((size_t)b * NCH + c) * D_INNER + d;
    #pragma unroll
    for (int n = 0; n < D_STATE; ++n) hIn[base * D_STATE + n] = h[n];
    const float ds = dtSum[base];
    #pragma unroll
    for (int n = 0; n < D_STATE; ++n)
      h[n] = fmaf(h[n], __expf(ds * Areg[n]), hEnd[base * D_STATE + n]);
  }
}

// Pass 3: rescan each chunk from correct h_in; y = h.C + D*x, gate SiLU(z).
__global__ __launch_bounds__(256) void k_scan_pass3(
    const float*  __restrict__ xz,      // (NTOK, 2*D_INNER): z at col 2048+d
    const __bf16* __restrict__ x_bf,
    const float*  __restrict__ x_dbl,   // B at 64..79, C at 80..95
    const float*  __restrict__ dt_raw,
    const float*  __restrict__ dt_bias,
    const float*  __restrict__ A_log,
    const float*  __restrict__ Dp,
    const float*  __restrict__ hIn,
    __bf16*       __restrict__ y_bf)    // (NTOK, D_INNER)
{
  const int d = blockIdx.x * 256 + threadIdx.x;
  const int c = blockIdx.y;
  const int b = blockIdx.z;

  float Areg[D_STATE];
  #pragma unroll
  for (int n = 0; n < D_STATE; ++n) Areg[n] = -__expf(A_log[d * D_STATE + n]);
  const float bias = dt_bias[d];
  const float Dd   = Dp[d];

  float h[D_STATE];
  const size_t base = ((size_t)b * NCH + c) * D_INNER + d;
  #pragma unroll
  for (int n = 0; n < D_STATE; ++n) h[n] = hIn[base * D_STATE + n];

  for (int l = c * CHL; l < (c + 1) * CHL; ++l) {
    const size_t t = (size_t)b * SEQ + l;
    float dt = softplusf(dt_raw[t * D_INNER + d] + bias);
    const float xv = (float)x_bf[t * D_INNER + d];
    const float dx = dt * xv;
    const float* BC = x_dbl + t * XPROJ_NP + 64;
    float y = 0.f;
    #pragma unroll
    for (int n = 0; n < D_STATE; ++n) {
      h[n] = fmaf(h[n], __expf(dt * Areg[n]), dx * BC[n]);
      y = fmaf(h[n], BC[16 + n], y);
    }
    y = fmaf(xv, Dd, y);                                 // + D*x skip
    const float z = xz[t * (size_t)(2 * D_INNER) + D_INNER + d];
    y *= z / (1.f + __expf(-z));                         // * SiLU(z)
    y_bf[t * D_INNER + d] = (__bf16)y;
  }
}

// ================= launcher ==================
extern "C" void kernel_launch(void* const* d_in, const int* in_sizes, int n_in,
                              void* d_out, int out_size, void* d_ws, size_t ws_size,
                              hipStream_t stream)
{
  (void)in_sizes; (void)n_in; (void)out_size; (void)ws_size;
  const float* hs    = (const float*)d_in[0];
  const float* w_in  = (const float*)d_in[1];
  const float* cw    = (const float*)d_in[2];
  const float* cb    = (const float*)d_in[3];
  const float* w_xp  = (const float*)d_in[4];
  const float* w_dt  = (const float*)d_in[5];
  const float* dt_b  = (const float*)d_in[6];
  const float* A_log = (const float*)d_in[7];
  const float* Dp    = (const float*)d_in[8];
  const float* w_out = (const float*)d_in[9];
  float* out = (float*)d_out;

  char* p = (char*)d_ws;
  auto alloc = [&](size_t bytes) { char* r = p; p += (bytes + 255) & ~(size_t)255; return r; };

  __bf16* hs_bf    = (__bf16*)alloc((size_t)NTOK * D_MODEL * 2);
  __bf16* w_in_bf  = (__bf16*)alloc((size_t)2 * D_INNER * D_MODEL * 2);
  __bf16* w_xp_bf  = (__bf16*)alloc((size_t)XPROJ_NP * D_INNER * 2);
  __bf16* w_dt_bf  = (__bf16*)alloc((size_t)D_INNER * DT_RANK * 2);
  __bf16* w_out_bf = (__bf16*)alloc((size_t)D_MODEL * D_INNER * 2);
  float*  xz       = (float*) alloc((size_t)NTOK * 2 * D_INNER * 4);
  __bf16* x_bf     = (__bf16*)alloc((size_t)NTOK * D_INNER * 2);
  float*  x_dbl    = (float*) alloc((size_t)NTOK * XPROJ_NP * 4);
  __bf16* dtin_bf  = (__bf16*)alloc((size_t)NTOK * DT_RANK * 2);
  float*  dt_raw   = (float*) alloc((size_t)NTOK * D_INNER * 4);
  __bf16* y_bf     = (__bf16*)alloc((size_t)NTOK * D_INNER * 2);
  float*  hEnd     = (float*) alloc((size_t)BATCH * NCH * D_INNER * D_STATE * 4);
  float*  hIn      = (float*) alloc((size_t)BATCH * NCH * D_INNER * D_STATE * 4);
  float*  dtSum    = (float*) alloc((size_t)BATCH * NCH * D_INNER * 4);

  auto cvt = [&](const float* src, __bf16* dst, size_t n) {
    k_f32_to_bf16<<<(unsigned)((n + 255) / 256), 256, 0, stream>>>(src, dst, n);
  };

  // 1) bf16 conversions
  cvt(hs,    hs_bf,    (size_t)NTOK * D_MODEL);
  cvt(w_in,  w_in_bf,  (size_t)2 * D_INNER * D_MODEL);
  cvt(w_dt,  w_dt_bf,  (size_t)D_INNER * DT_RANK);
  cvt(w_out, w_out_bf, (size_t)D_MODEL * D_INNER);
  k_cvt_pad_xproj<<<(XPROJ_NP * D_INNER) / 256, 256, 0, stream>>>(w_xp, w_xp_bf);

  // 2) in_proj: xz (16384 x 4096) = hs (16384 x 1024) * w_in^T
  k_gemm_bf16_wmma<<<dim3((2 * D_INNER) / TN, NTOK / TM), 256, 0, stream>>>(
      hs_bf, w_in_bf, xz, NTOK, 2 * D_INNER, D_MODEL);

  // 3) causal conv + SiLU -> x (bf16)
  k_conv_silu<<<(unsigned)(((size_t)NTOK * D_INNER) / 256), 256, 0, stream>>>(
      xz, cw, cb, x_bf);

  // 4) x_proj: x_dbl (16384 x 128pad) = x * w_xp^T
  k_gemm_bf16_wmma<<<dim3(XPROJ_NP / TN, NTOK / TM), 256, 0, stream>>>(
      x_bf, w_xp_bf, x_dbl, NTOK, XPROJ_NP, D_INNER);

  // 5) dt_proj: dt_raw (16384 x 2048) = dt_in (16384 x 64) * w_dt^T
  k_extract_dtin<<<(unsigned)(((size_t)NTOK * DT_RANK) / 256), 256, 0, stream>>>(
      x_dbl, dtin_bf);
  k_gemm_bf16_wmma<<<dim3(D_INNER / TN, NTOK / TM), 256, 0, stream>>>(
      dtin_bf, w_dt_bf, dt_raw, NTOK, D_INNER, DT_RANK);

  // 6) chunked selective scan (16x parallel over L) -> y (bf16)
  k_scan_pass1<<<dim3(D_INNER / 256, NCH, BATCH), 256, 0, stream>>>(
      x_bf, x_dbl, dt_raw, dt_b, A_log, hEnd, dtSum);
  k_scan_pass2<<<(BATCH * D_INNER) / 256, 256, 0, stream>>>(
      A_log, hEnd, dtSum, hIn);
  k_scan_pass3<<<dim3(D_INNER / 256, NCH, BATCH), 256, 0, stream>>>(
      xz, x_bf, x_dbl, dt_raw, dt_b, A_log, Dp, hIn, y_bf);

  // 7) out_proj: out (16384 x 1024) = y * w_out^T  (f32 to d_out)
  k_gemm_bf16_wmma<<<dim3(D_MODEL / TN, NTOK / TM), 256, 0, stream>>>(
      y_bf, w_out_bf, out, NTOK, D_MODEL, D_INNER);
}
